// RecursiveNN_56710748176690
// MI455X (gfx1250) — matrix operand order repaired and checked by reading, not compiled, not run on hardware
//
#include <hip/hip_runtime.h>

// RecursiveNN tree contraction for MI455X (gfx1250, wave32).
//   x = embedding[word_ids]; 11x: x = (B, n/2, 256) @ W.T + b  -> (256, 128)
//
// One WG per batch element; whole tree reduction lives in LDS (zero HBM
// intermediates; gather of 256 MB is the memory floor, ~11us at 23.3 TB/s).
// Matmul core: split-f16 emulated fp32 on V_WMMA_F32_16X16X32_F16:
//   x = hi + lo (f16 pair), x.w = ah*bh + ah*bl + al*bh  (error ~2^-21/level)
// 3 WMMAs per 32-K block vs 64 K4-f32 WMMAs per group -> far less issue
// pressure; fp32 accumulation keeps accuracy near-fp32 (~1e-5 end to end).
// Waves tiled 2(M) x 4(N, 2 tiles each) to halve redundant LDS A reads.
// Gather is register-prefetched one chunk ahead to overlap with WMMA levels.

typedef _Float16 f16;
typedef _Float16 v4h  __attribute__((ext_vector_type(4)));
typedef _Float16 v8h  __attribute__((ext_vector_type(8)));
typedef _Float16 v16h __attribute__((ext_vector_type(16)));
typedef float    v8f  __attribute__((ext_vector_type(8)));

#define W_H     264   // W row stride (halves): 256 + 8 pad -> 33 16B-granules (odd) => conflict-free
#define NODE_H  136   // node stride (halves): 128 + 8 pad, keeps 16B alignment across pair boundary
#define XP_H    272   // concatenated-pair row stride = 2 * NODE_H

// LDS regions (offsets in halves); every size is a multiple of 8 halves -> 16B aligned
#define WH_OFF  0                         // W hi: 128 x 264
#define WL_OFF  (WH_OFF + 128 * W_H)      // W lo
#define XH_OFF  (WL_OFF + 128 * W_H)      // bufX hi: 128 nodes (leaf chunk)
#define XL_OFF  (XH_OFF + 128 * NODE_H)
#define YH_OFF  (XL_OFF + 128 * NODE_H)   // bufY: 64 nodes
#define YL_OFF  (YH_OFF + 64 * NODE_H)
#define ZH_OFF  (YL_OFF + 64 * NODE_H)    // bufZ: 32 nodes
#define ZL_OFF  (ZH_OFF + 32 * NODE_H)
#define SH_OFF  (ZL_OFF + 32 * NODE_H)    // stage: 16 chunk results
#define SL_OFF  (SH_OFF + 16 * NODE_H)
#define SMEM_H  (SL_OFF + 16 * NODE_H)    // 132864 halves = 265728 B < 320 KB

extern __shared__ f16 hmem[];

__device__ __forceinline__ v16h cat8(v8h a, v8h b) {
    return __builtin_shufflevector(a, b, 0,1,2,3,4,5,6,7,8,9,10,11,12,13,14,15);
}

struct HL { f16 h; f16 l; };
__device__ __forceinline__ HL split_f32(float v) {
    HL r;
    r.h = (f16)v;
    r.l = (f16)(v - (float)r.h);
    return r;
}

// One tree level: out[m][n] = sum_k in_pairs[m][k]*W[n][k] + b[n], m < n_out.
// Wave (mw, nbase) covers M-groups mt = mw, mw+2, ... and channels [nbase, nbase+32).
__device__ __forceinline__ void level_step(const f16* Ah, const f16* Al,
                                           f16* Oh, f16* Ol, int n_out,
                                           const f16* Wh, const f16* Wl,
                                           float bias0, float bias1,
                                           int lane, int hiL, int mw, int nbase) {
    const int n_groups = (n_out + 15) >> 4;
    const int n0 = nbase + (lane & 15);
    const int n1 = n0 + 16;
    const f16* wh0 = Wh + n0 * W_H;  const f16* wl0 = Wl + n0 * W_H;
    const f16* wh1 = Wh + n1 * W_H;  const f16* wl1 = Wl + n1 * W_H;
    const int koffA = hiL * 8;       // A: lane half selects K {0..7,16..23} vs {8..15,24..31}
    const int koffB = hiL * 8;       // B: lane half selects K {0..7,16..23} vs {8..15,24..31}

    for (int mt = mw; mt < n_groups; mt += 2) {
        int m = mt * 16 + (lane & 15);
        int mi = (m < n_out) ? m : 0;         // clamp: garbage rows are never stored;
                                              // EXEC stays all-ones for every WMMA
        const f16* arh = Ah + mi * XP_H;
        const f16* arl = Al + mi * XP_H;

        v8f c0 = (v8f)0.0f, c1 = (v8f)0.0f;
        #pragma unroll 2
        for (int kb = 0; kb < 8; ++kb) {
            int k0 = kb * 32;
            int ka = k0 + ((k0 >= 128) ? 8 : 0) + koffA;  // skip pad between the node pair
            v16h aH = cat8(*(const v8h*)(arh + ka), *(const v8h*)(arh + ka + 16));
            v16h aL = cat8(*(const v8h*)(arl + ka), *(const v8h*)(arl + ka + 16));
            int kw = k0 + koffB;
            v16h bH0 = cat8(*(const v8h*)(wh0 + kw), *(const v8h*)(wh0 + kw + 16));
            v16h bL0 = cat8(*(const v8h*)(wl0 + kw), *(const v8h*)(wl0 + kw + 16));
            v16h bH1 = cat8(*(const v8h*)(wh1 + kw), *(const v8h*)(wh1 + kw + 16));
            v16h bL1 = cat8(*(const v8h*)(wl1 + kw), *(const v8h*)(wl1 + kw + 16));
            // split-f32 product: hi*hi + hi*lo + lo*hi (lo*lo ~ 2^-22, dropped)
            c0 = __builtin_amdgcn_wmma_f32_16x16x32_f16(false, aH, false, bH0, (short)0, c0, false, false);
            c0 = __builtin_amdgcn_wmma_f32_16x16x32_f16(false, aH, false, bL0, (short)0, c0, false, false);
            c0 = __builtin_amdgcn_wmma_f32_16x16x32_f16(false, aL, false, bH0, (short)0, c0, false, false);
            c1 = __builtin_amdgcn_wmma_f32_16x16x32_f16(false, aH, false, bH1, (short)0, c1, false, false);
            c1 = __builtin_amdgcn_wmma_f32_16x16x32_f16(false, aH, false, bL1, (short)0, c1, false, false);
            c1 = __builtin_amdgcn_wmma_f32_16x16x32_f16(false, aL, false, bH1, (short)0, c1, false, false);
        }

        // C/D layout: VGPR j -> M = j + 8*hiL, N = lane%16 (per N-tile)
        #pragma unroll
        for (int j = 0; j < 8; ++j) {
            int mrow = mt * 16 + j + hiL * 8;
            if (mrow < n_out) {
                HL r0 = split_f32(c0[j] + bias0);
                Oh[mrow * NODE_H + n0] = r0.h;  Ol[mrow * NODE_H + n0] = r0.l;
                HL r1 = split_f32(c1[j] + bias1);
                Oh[mrow * NODE_H + n1] = r1.h;  Ol[mrow * NODE_H + n1] = r1.l;
            }
        }
    }
    __syncthreads();
}

// Gather pipeline: issue 16 b128 global loads for half an embedding row.
__device__ __forceinline__ void gather_issue(const int* wid_row, const float* emb,
                                             int chunk, int row, int part, float4* pf) {
    int wid = wid_row[chunk * 128 + row];
    const float4* src = (const float4*)(emb + (long)wid * 128) + part * 16;
    #pragma unroll
    for (int j = 0; j < 16; ++j) pf[j] = src[j];
}
__device__ __forceinline__ void gather_store(f16* Xh, f16* Xl,
                                             int row, int part, const float4* pf) {
    f16* dh = Xh + row * NODE_H + part * 64;
    f16* dl = Xl + row * NODE_H + part * 64;
    #pragma unroll
    for (int j = 0; j < 16; ++j) {
        float4 v = pf[j];
        HL r0 = split_f32(v.x), r1 = split_f32(v.y);
        HL r2 = split_f32(v.z), r3 = split_f32(v.w);
        v4h h, l;
        h[0] = r0.h; h[1] = r1.h; h[2] = r2.h; h[3] = r3.h;
        l[0] = r0.l; l[1] = r1.l; l[2] = r2.l; l[3] = r3.l;
        *(v4h*)(dh + j * 4) = h;
        *(v4h*)(dl + j * 4) = l;
    }
}

__global__ __launch_bounds__(256, 1)
void recursive_nn_tree(const int* __restrict__ word_ids,
                       const float* __restrict__ emb,
                       const float* __restrict__ W,
                       const float* __restrict__ bvec,
                       float* __restrict__ out) {
    const int b    = blockIdx.x;
    const int tid  = threadIdx.x;       // 256 threads = 8 wave32s
    const int lane = tid & 31;
    const int wave = tid >> 5;
    const int hiL  = lane >> 4;
    const int mw   = wave & 1;          // M-half owned by this wave
    const int nb   = (wave >> 1) * 32;  // 32-channel N range (2 WMMA tiles)

    f16* Wh = hmem + WH_OFF;  f16* Wl = hmem + WL_OFF;
    f16* Xh = hmem + XH_OFF;  f16* Xl = hmem + XL_OFF;
    f16* Yh = hmem + YH_OFF;  f16* Yl = hmem + YL_OFF;
    f16* Zh = hmem + ZH_OFF;  f16* Zl = hmem + ZL_OFF;
    f16* Sh = hmem + SH_OFF;  f16* Sl = hmem + SL_OFF;

    const int row = tid >> 1, part = tid & 1;   // 2 threads per 128-float row

    // Stage W (128x256 f32) into LDS as split-f16 hi/lo, padded rows.
    {
        const float4* ws = (const float4*)(W + row * 256) + part * 16;
        f16* dh = Wh + row * W_H + part * 64;
        f16* dl = Wl + row * W_H + part * 64;
        #pragma unroll
        for (int j = 0; j < 16; ++j) {
            float4 v = ws[j];
            HL r0 = split_f32(v.x), r1 = split_f32(v.y);
            HL r2 = split_f32(v.z), r3 = split_f32(v.w);
            v4h h, l;
            h[0] = r0.h; h[1] = r1.h; h[2] = r2.h; h[3] = r3.h;
            l[0] = r0.l; l[1] = r1.l; l[2] = r2.l; l[3] = r3.l;
            *(v4h*)(dh + j * 4) = h;
            *(v4h*)(dl + j * 4) = l;
        }
    }
    const float bias0 = bvec[nb + (lane & 15)];
    const float bias1 = bvec[nb + 16 + (lane & 15)];
    const int* wid_row = word_ids + b * 2048;

    // Prologue: gather chunk 0.
    float4 pf[16];
    gather_issue(wid_row, emb, 0, row, part, pf);
    gather_store(Xh, Xl, row, part, pf);
    __syncthreads();

    // 16 chunks of 128 leaves; 7 WMMA levels per chunk (128 -> 1).
    for (int chunk = 0; chunk < 16; ++chunk) {
        bool more = (chunk < 15);
        if (more) gather_issue(wid_row, emb, chunk + 1, row, part, pf);  // overlaps level 1

        level_step(Xh, Xl, Yh, Yl, 64, Wh, Wl, bias0, bias1, lane, hiL, mw, nb); // 128->64
        if (more) gather_store(Xh, Xl, row, part, pf);   // bufX free after level-1 sync
        level_step(Yh, Yl, Zh, Zl, 32, Wh, Wl, bias0, bias1, lane, hiL, mw, nb); //  64->32
        level_step(Zh, Zl, Yh, Yl, 16, Wh, Wl, bias0, bias1, lane, hiL, mw, nb); //  32->16
        level_step(Yh, Yl, Zh, Zl,  8, Wh, Wl, bias0, bias1, lane, hiL, mw, nb); //  16-> 8
        level_step(Zh, Zl, Yh, Yl,  4, Wh, Wl, bias0, bias1, lane, hiL, mw, nb); //   8-> 4
        level_step(Yh, Yl, Zh, Zl,  2, Wh, Wl, bias0, bias1, lane, hiL, mw, nb); //   4-> 2
        level_step(Zh, Zl, Yh, Yl,  1, Wh, Wl, bias0, bias1, lane, hiL, mw, nb); //   2-> 1

        if (tid < 128) {                       // chunk result -> stage slot
            Sh[chunk * NODE_H + tid] = Yh[tid];
            Sl[chunk * NODE_H + tid] = Yl[tid];
        }
        __syncthreads();
    }

    // Final 4 levels over 16 chunk results: 16 -> 8 -> 4 -> 2 -> 1.
    level_step(Sh, Sl, Zh, Zl, 8, Wh, Wl, bias0, bias1, lane, hiL, mw, nb);
    level_step(Zh, Zl, Yh, Yl, 4, Wh, Wl, bias0, bias1, lane, hiL, mw, nb);
    level_step(Yh, Yl, Zh, Zl, 2, Wh, Wl, bias0, bias1, lane, hiL, mw, nb);
    level_step(Zh, Zl, Yh, Yl, 1, Wh, Wl, bias0, bias1, lane, hiL, mw, nb);

    if (tid < 128)
        out[b * 128 + tid] = (float)Yh[tid] + (float)Yl[tid];  // reconstruct fp32
}

extern "C" void kernel_launch(void* const* d_in, const int* in_sizes, int n_in,
                              void* d_out, int out_size, void* d_ws, size_t ws_size,
                              hipStream_t stream) {
    (void)in_sizes; (void)n_in; (void)out_size; (void)d_ws; (void)ws_size;
    const int*   word_ids = (const int*)d_in[0];    // (256, 2048) int32
    const float* emb      = (const float*)d_in[1];  // (100000, 128) f32
    const float* W        = (const float*)d_in[2];  // (128, 256) f32
    const float* bvec     = (const float*)d_in[3];  // (128,) f32
    float*       outp     = (float*)d_out;          // (256, 128) f32

    recursive_nn_tree<<<256, 256, SMEM_H * sizeof(f16), stream>>>(
        word_ids, emb, W, bvec, outp);
}
